// Model_15951508538248
// MI455X (gfx1250) — matrix operand order repaired and checked by reading, not compiled, not run on hardware
//
#include <hip/hip_runtime.h>
#include <hip/hip_bf16.h>
#include <math.h>

// ---------------- types ----------------
typedef __attribute__((ext_vector_type(16))) __bf16 v16bf;
typedef __attribute__((ext_vector_type(8)))  float  v8f;
typedef __attribute__((ext_vector_type(4)))  unsigned int v4u;
typedef __attribute__((ext_vector_type(8)))  int v8i_t;
typedef __attribute__((ext_vector_type(4)))  int v4i_t;

union BF16x16 {
    v16bf v;
    unsigned short u[16];
    unsigned int   u32[8];
    uint4          q[2];
};

#define TOK   8192      // B*N tokens
#define EDIM  1024
#define NPTS  2048
#define BB    4
#define NH    4
#define HD    256
#define QKVN  3072

#if __has_builtin(__builtin_amdgcn_tensor_load_to_lds) && __has_builtin(__builtin_amdgcn_s_wait_tensorcnt)
#define USE_TDM 1
#else
#define USE_TDM 0
#endif

__device__ __forceinline__ unsigned short f2bf(float f) {
    unsigned int u = __float_as_uint(f);
    unsigned int r = u + 0x7fffu + ((u >> 16) & 1u);
    return (unsigned short)(r >> 16);
}

__device__ __forceinline__ v8f v8f_zero() {
    v8f z;
#pragma unroll
    for (int i = 0; i < 8; ++i) z[i] = 0.0f;
    return z;
}

// A-fragment K offset pattern for 16-bit 16x32 A matrix (contiguous order: k = 16*half + 2v)
__device__ __forceinline__ int a_koff(int v, int half) {
    return v * 2 + half * 16;
}

// ---------------- conversions ----------------
__global__ void cvt_f32_bf16(const float* __restrict__ in, unsigned short* __restrict__ out, int n) {
    int i = blockIdx.x * blockDim.x + threadIdx.x;
    if (i < n) out[i] = f2bf(in[i]);
}

// features (B,256,N) -> token-major bf16 [t][c]
__global__ void feat_to_tok(const float* __restrict__ f, unsigned short* __restrict__ out) {
    int i = blockIdx.x * blockDim.x + threadIdx.x;   // over TOK*256
    if (i >= TOK * 256) return;
    int c = i & 255;
    int t = i >> 8;
    int b = t >> 11;
    int n = t & 2047;
    out[(size_t)t * 256 + c] = f2bf(f[((size_t)(b * 256 + c)) * 2048 + n]);
}

__global__ void add_cvt(const float* __restrict__ a, const float* __restrict__ b,
                        float* __restrict__ of, unsigned short* __restrict__ ob, int n) {
    int i = blockIdx.x * blockDim.x + threadIdx.x;
    if (i >= n) return;
    float v = a[i] + b[i];
    of[i] = v;
    ob[i] = f2bf(v);
}

__global__ void zero_f(float* p, int n) {
    int i = blockIdx.x * blockDim.x + threadIdx.x;
    if (i < n) p[i] = 0.0f;
}

// ---------------- WMMA GEMM ----------------
// C[t][j] = act( sum_k A[t][k] * W[j][k] + bias[j] )
// A: M x K bf16 row-major (tokens), W: N x K bf16 row-major (weights)
// One wave computes a 16x64 tile (4 N-tiles sharing one A fragment);
// 8 waves (256 threads) per block. N must be a multiple of 64.
__global__ __launch_bounds__(256) void gemm_wmma(
    const unsigned short* __restrict__ A,
    const unsigned short* __restrict__ W,
    const float* __restrict__ bias,
    float* __restrict__ Cf,
    unsigned short* __restrict__ Cb,
    int M, int N, int K, int act)
{
    int wave = threadIdx.x >> 5;
    int lane = threadIdx.x & 31;
    int half = lane >> 4;
    int ln   = lane & 15;
    int ntg  = N >> 6;                      // groups of 4 N-tiles
    int gidx = blockIdx.x * 8 + wave;
    int tg = gidx % ntg;
    int tm = gidx / ntg;
    if (tm >= (M >> 4)) return;

    const unsigned short* arow = A + (size_t)(tm * 16 + ln) * K;
    const unsigned short* wrow0 = W + (size_t)(tg * 64 + ln) * K + half * 16;

    v8f acc[4];
#pragma unroll
    for (int j = 0; j < 4; ++j) acc[j] = v8f_zero();

    for (int k0 = 0; k0 < K; k0 += 32) {
        BF16x16 a;
#pragma unroll
        for (int v = 0; v < 8; ++v)
            a.u32[v] = *(const unsigned int*)(arow + k0 + a_koff(v, half));
#pragma unroll
        for (int j = 0; j < 4; ++j) {
            BF16x16 b;
            const uint4* wp = (const uint4*)(wrow0 + (size_t)(j * 16) * K + k0);
            b.q[0] = wp[0];
            b.q[1] = wp[1];
            acc[j] = __builtin_amdgcn_wmma_f32_16x16x32_bf16(false, a.v, false, b.v,
                                                             (short)0, acc[j], false, false);
        }
    }

#pragma unroll
    for (int j = 0; j < 4; ++j) {
        int col = tg * 64 + j * 16 + ln;
        float bv = bias ? bias[col] : 0.0f;
#pragma unroll
        for (int r = 0; r < 8; ++r) {
            int row = tm * 16 + r + half * 8;
            float v = acc[j][r] + bv;
            if (act == 1) v = 0.5f * v * (1.0f + erff(v * 0.70710678118654752f));
            size_t o = (size_t)row * N + col;
            if (Cf) Cf[o] = v;
            if (Cb) Cb[o] = f2bf(v);
        }
    }
}

// ---------------- LayerNorm (per token over 1024) ----------------
__device__ __forceinline__ float block_sum256(float v, float* sred) {
#pragma unroll
    for (int m = 1; m < 32; m <<= 1) v += __shfl_xor(v, m, 32);
    int w = threadIdx.x >> 5, l = threadIdx.x & 31;
    if (l == 0) sred[w] = v;
    __syncthreads();
    float tot = 0.0f;
#pragma unroll
    for (int i = 0; i < 8; ++i) tot += sred[i];
    __syncthreads();
    return tot;
}

__device__ __forceinline__ float block_max256(float v, float* sred) {
#pragma unroll
    for (int m = 1; m < 32; m <<= 1) v = fmaxf(v, __shfl_xor(v, m, 32));
    int w = threadIdx.x >> 5, l = threadIdx.x & 31;
    if (l == 0) sred[w] = v;
    __syncthreads();
    float tot = -1e30f;
#pragma unroll
    for (int i = 0; i < 8; ++i) tot = fmaxf(tot, sred[i]);
    __syncthreads();
    return tot;
}

__global__ __launch_bounds__(256) void ln_kernel(
    const float* __restrict__ X, const float* __restrict__ R,
    const float* __restrict__ g, const float* __restrict__ bta,
    float* __restrict__ Yf, unsigned short* __restrict__ Yb)
{
    __shared__ float sred[8];
    int t = blockIdx.x;
    int tid = threadIdx.x;
    const float* xr = X + (size_t)t * EDIM;
    const float* rr = R ? (R + (size_t)t * EDIM) : nullptr;
    float v[4];
    float s = 0.0f;
#pragma unroll
    for (int i = 0; i < 4; ++i) {
        int c = tid + i * 256;
        v[i] = xr[c] + (rr ? rr[c] : 0.0f);
        s += v[i];
    }
    float mean = block_sum256(s, sred) * (1.0f / 1024.0f);
    float var = 0.0f;
#pragma unroll
    for (int i = 0; i < 4; ++i) { float d = v[i] - mean; var += d * d; }
    var = block_sum256(var, sred) * (1.0f / 1024.0f);
    float inv = rsqrtf(var + 1e-5f);
#pragma unroll
    for (int i = 0; i < 4; ++i) {
        int c = tid + i * 256;
        float y = (v[i] - mean) * inv * g[c] + bta[c];
        size_t o = (size_t)t * EDIM + c;
        if (Yf) Yf[o] = y;
        if (Yb) Yb[o] = f2bf(y);
    }
}

// ---------------- Flash attention (WMMA + TDM V staging) ----------------
// qkv: [t][3072] bf16 (q|k|v), per head hd=256. One wave per (b, h, 16-query block).
__global__ __launch_bounds__(32) void attn_wmma(
    const unsigned short* __restrict__ qkv, unsigned short* __restrict__ obf)
{
    __shared__ __align__(16) unsigned short plds[16 * 32];
    __shared__ __align__(16) unsigned short vlds[32 * 256];

    int qb = blockIdx.x, h = blockIdx.y, b = blockIdx.z;
    int lane = threadIdx.x;
    int half = lane >> 4;
    int ln   = lane & 15;
    const size_t RS = QKVN;
    const unsigned short* Qb = qkv + (size_t)b * 2048 * RS + h * HD;
    const unsigned short* Kb = Qb + 1024;
    const unsigned short* Vb = Qb + 2048;

    // resident Q fragments: 8 chunks along d
    BF16x16 qf[8];
    const unsigned short* qrow = Qb + (size_t)(qb * 16 + ln) * RS;
#pragma unroll
    for (int kc = 0; kc < 8; ++kc) {
#pragma unroll
        for (int v = 0; v < 8; ++v)
            qf[kc].u32[v] = *(const unsigned int*)(qrow + kc * 32 + a_koff(v, half));
    }

    v8f oacc[16];
#pragma unroll
    for (int dc = 0; dc < 16; ++dc) oacc[dc] = v8f_zero();
    float mrow[8], lrow[8];
#pragma unroll
    for (int r = 0; r < 8; ++r) { mrow[r] = -1e30f; lrow[r] = 0.0f; }

#if USE_TDM
    unsigned lds_off = (unsigned)(size_t)(void*)vlds;   // LDS aperture: low 32 bits = LDS offset
#endif

    for (int mc = 0; mc < 2048; mc += 32) {
        // ---- stage V chunk (32 keys x 256 d, bf16, row stride 3072) into LDS ----
#if USE_TDM
        {
            unsigned long long ga = (unsigned long long)(const void*)(Vb + (size_t)mc * RS);
            v4u g0;
            g0[0] = 1u;                                   // count=1, user descriptor
            g0[1] = lds_off;                              // lds_addr
            g0[2] = (unsigned)ga;                         // global_addr[31:0]
            g0[3] = (unsigned)(ga >> 32) | (2u << 30);    // global_addr[56:32], type=2 (image)
            v8i_t g1;
            g1[0] = (int)(1u << 16);                      // data_size=1 -> 2 bytes
            g1[1] = (int)(256u << 16);                    // tensor_dim0[15:0]=256 in bits[31:16]
            g1[2] = (int)(32u << 16);                     // tensor_dim0[31:16]=0, tensor_dim1[15:0]=32
            g1[3] = (int)(256u << 16);                    // tensor_dim1[31:16]=0, tile_dim0=256
            g1[4] = (int)32u;                             // tile_dim1=32, tile_dim2=0
            g1[5] = (int)3072;                            // tensor_dim0_stride[31:0]
            g1[6] = 0;                                    // stride hi bits, dim1_stride lo
            g1[7] = 0;
            v4i_t z4 = {0, 0, 0, 0};
            v8i_t z8 = {0, 0, 0, 0, 0, 0, 0, 0};
            __builtin_amdgcn_tensor_load_to_lds(g0, g1, z4, z4, z8, 0);
        }
#else
        {
            const uint4* src = (const uint4*)(Vb + (size_t)(mc + lane) * RS);
            uint4* dst = (uint4*)(vlds + lane * 256);
#pragma unroll
            for (int s2 = 0; s2 < 32; ++s2) dst[s2] = src[s2];
        }
#endif

        // ---- scores: two 16x16 tiles over d (K=256) ----
        v8f s0 = v8f_zero(), s1 = v8f_zero();
#pragma unroll
        for (int kc = 0; kc < 8; ++kc) {
            BF16x16 b0, b1;
            const uint4* k0p = (const uint4*)(Kb + (size_t)(mc + ln) * RS + kc * 32 + half * 16);
            b0.q[0] = k0p[0]; b0.q[1] = k0p[1];
            const uint4* k1p = (const uint4*)(Kb + (size_t)(mc + 16 + ln) * RS + kc * 32 + half * 16);
            b1.q[0] = k1p[0]; b1.q[1] = k1p[1];
            s0 = __builtin_amdgcn_wmma_f32_16x16x32_bf16(false, qf[kc].v, false, b0.v,
                                                         (short)0, s0, false, false);
            s1 = __builtin_amdgcn_wmma_f32_16x16x32_bf16(false, qf[kc].v, false, b1.v,
                                                         (short)0, s1, false, false);
        }

        // ---- online softmax; rows r+8*half live across the 16 lanes of each half ----
        float corr[8];
#pragma unroll
        for (int r = 0; r < 8; ++r) {
            float a0 = s0[r] * 0.0625f;          // 1/sqrt(256)
            float a1 = s1[r] * 0.0625f;
            float mx = fmaxf(a0, a1);
            mx = fmaxf(mx, __shfl_xor(mx, 1, 32));
            mx = fmaxf(mx, __shfl_xor(mx, 2, 32));
            mx = fmaxf(mx, __shfl_xor(mx, 4, 32));
            mx = fmaxf(mx, __shfl_xor(mx, 8, 32));
            float mnew = fmaxf(mrow[r], mx);
            float c = __expf(mrow[r] - mnew);
            float p0 = __expf(a0 - mnew);
            float p1 = __expf(a1 - mnew);
            float ls = p0 + p1;
            ls += __shfl_xor(ls, 1, 32);
            ls += __shfl_xor(ls, 2, 32);
            ls += __shfl_xor(ls, 4, 32);
            ls += __shfl_xor(ls, 8, 32);
            lrow[r] = lrow[r] * c + ls;
            mrow[r] = mnew;
            corr[r] = c;
            int prow = r + half * 8;
            plds[prow * 32 + ln]      = f2bf(p0);
            plds[prow * 32 + 16 + ln] = f2bf(p1);
        }
        __syncthreads();

        // P as A-fragment (16x32 bf16)
        BF16x16 pf;
#pragma unroll
        for (int v = 0; v < 8; ++v)
            pf.u32[v] = *(const unsigned int*)(plds + ln * 32 + a_koff(v, half));

#if USE_TDM
        __builtin_amdgcn_s_wait_tensorcnt(0);   // V tile resident in LDS
#endif

        // ---- O += P * V, 16 d-chunks of 16 ----
#pragma unroll
        for (int dc = 0; dc < 16; ++dc) {
            BF16x16 vf;
#pragma unroll
            for (int i = 0; i < 16; ++i)
                vf.u[i] = vlds[(half * 16 + i) * 256 + dc * 16 + ln];
#pragma unroll
            for (int r = 0; r < 8; ++r) oacc[dc][r] *= corr[r];
            oacc[dc] = __builtin_amdgcn_wmma_f32_16x16x32_bf16(false, pf.v, false, vf.v,
                                                               (short)0, oacc[dc], false, false);
        }
        __syncthreads();
    }

    // normalize + store bf16 into [t][1024]
#pragma unroll
    for (int dc = 0; dc < 16; ++dc) {
#pragma unroll
        for (int r = 0; r < 8; ++r) {
            float val = oacc[dc][r] / lrow[r];
            size_t t = (size_t)b * 2048 + qb * 16 + r + half * 8;
            obf[t * EDIM + h * HD + dc * 16 + ln] = f2bf(val);
        }
    }
}

// ---------------- softmax over N + pred = prob @ x ----------------
__global__ __launch_bounds__(256) void softmax_pred(
    const float* __restrict__ S, const float* __restrict__ x,
    float* __restrict__ predws, float* __restrict__ outpred)
{
    __shared__ float sred[8];
    int be = blockIdx.x;
    int b = be >> 10;
    int e = be & 1023;
    int tid = threadIdx.x;
    const float* base = S + ((size_t)b * 2048) * EDIM + e;
    float mx = -1e30f;
    for (int n = tid; n < 2048; n += 256) mx = fmaxf(mx, base[(size_t)n * EDIM]);
    mx = block_max256(mx, sred);
    float sum = 0.0f, p0 = 0.0f, p1 = 0.0f, p2 = 0.0f;
    const float* xb = x + (size_t)b * 2048 * 3;
    for (int n = tid; n < 2048; n += 256) {
        float w = __expf(base[(size_t)n * EDIM] - mx);
        sum += w;
        p0 += w * xb[n * 3 + 0];
        p1 += w * xb[n * 3 + 1];
        p2 += w * xb[n * 3 + 2];
    }
    sum = block_sum256(sum, sred);
    p0 = block_sum256(p0, sred);
    p1 = block_sum256(p1, sred);
    p2 = block_sum256(p2, sred);
    if (tid == 0) {
        float inv = 1.0f / sum;
        size_t o = (size_t)be * 3;
        float q0 = p0 * inv, q1 = p1 * inv, q2 = p2 * inv;
        predws[o] = q0; predws[o + 1] = q1; predws[o + 2] = q2;
        outpred[o] = q0; outpred[o + 1] = q1; outpred[o + 2] = q2;
    }
}

// ---------------- Chamfer ----------------
__global__ void chamfer_min(const float* __restrict__ P, const float* __restrict__ G,
                            int Np, int Ng, float* __restrict__ acc, float scale)
{
    int i = blockIdx.x * blockDim.x + threadIdx.x;
    if (i >= BB * Np) return;
    int b = i / Np;
    float px = P[(size_t)i * 3 + 0];
    float py = P[(size_t)i * 3 + 1];
    float pz = P[(size_t)i * 3 + 2];
    const float* g = G + (size_t)b * Ng * 3;
    float best = 1e30f;
    for (int j = 0; j < Ng; ++j) {
        float dx = px - g[j * 3 + 0];
        float dy = py - g[j * 3 + 1];
        float dz = pz - g[j * 3 + 2];
        float d = dx * dx + dy * dy + dz * dz;
        best = fminf(best, d);
    }
    atomicAdd(&acc[b], best * scale);
}

// ---------------- 10-NN over pred (per batch, 1024 points) ----------------
__global__ void knn_kernel(const float* __restrict__ P, int* __restrict__ idx)
{
    int i = blockIdx.x * blockDim.x + threadIdx.x;
    if (i >= BB * 1024) return;
    int b = i >> 10;
    const float* pb = P + (size_t)b * 1024 * 3;
    float px = P[(size_t)i * 3 + 0];
    float py = P[(size_t)i * 3 + 1];
    float pz = P[(size_t)i * 3 + 2];
    float bd[10];
    int   bi[10];
#pragma unroll
    for (int k = 0; k < 10; ++k) { bd[k] = 1e30f; bi[k] = 0; }
    for (int j = 0; j < 1024; ++j) {
        float dx = px - pb[j * 3 + 0];
        float dy = py - pb[j * 3 + 1];
        float dz = pz - pb[j * 3 + 2];
        float d = dx * dx + dy * dy + dz * dz;
        if (d < bd[9]) {
            int k = 9;
            while (k > 0 && bd[k - 1] > d) { bd[k] = bd[k - 1]; bi[k] = bi[k - 1]; --k; }
            bd[k] = d; bi[k] = j;
        }
    }
#pragma unroll
    for (int k = 0; k < 10; ++k) idx[(size_t)i * 10 + k] = bi[k];
}

// ---------------- neighbor loss accumulation ----------------
__global__ void neighbor_kernel(const float* __restrict__ P, const int* __restrict__ idx,
                                float* __restrict__ pp /* [bb*4+b] */)
{
    int i = blockIdx.x * blockDim.x + threadIdx.x;
    if (i >= BB * 1024) return;
    int b = i >> 10;
    int p = i & 1023;
    float sx = P[((size_t)b * 1024 + p) * 3 + 0];
    float sy = P[((size_t)b * 1024 + p) * 3 + 1];
    float sz = P[((size_t)b * 1024 + p) * 3 + 2];
    float local[16];
#pragma unroll
    for (int t = 0; t < 16; ++t) local[t] = 0.0f;
    for (int k = 1; k < 10; ++k) {
        int nb = idx[(size_t)i * 10 + k];
        float nx = P[((size_t)b * 1024 + nb) * 3 + 0] - sx;
        float ny = P[((size_t)b * 1024 + nb) * 3 + 1] - sy;
        float nz = P[((size_t)b * 1024 + nb) * 3 + 2] - sz;
        float srcsq = nx * nx + ny * ny + nz * nz;
        float g = __expf(-srcsq * 0.125f);   // 1/HEAT_T
#pragma unroll
        for (int bb = 0; bb < BB; ++bb) {
            float ax = P[((size_t)bb * 1024 + nb) * 3 + 0] - P[((size_t)bb * 1024 + p) * 3 + 0];
            float ay = P[((size_t)bb * 1024 + nb) * 3 + 1] - P[((size_t)bb * 1024 + p) * 3 + 1];
            float az = P[((size_t)bb * 1024 + nb) * 3 + 2] - P[((size_t)bb * 1024 + p) * 3 + 2];
            local[bb * 4 + b] += g * (ax * ax + ay * ay + az * az);
        }
    }
#pragma unroll
    for (int t = 0; t < 16; ++t)
        if (local[t] != 0.0f) atomicAdd(&pp[t], local[t]);
}

__global__ void finalize_kernel(const float* __restrict__ accCd, const float* __restrict__ pp,
                                float* __restrict__ out)
{
    if (threadIdx.x == 0 && blockIdx.x == 0) {
        float cd = 0.0f;
        for (int b = 0; b < BB; ++b) cd += accCd[b];
        cd *= 0.25f;
        float nl = 0.0f;
        for (int bb = 0; bb < BB; ++bb)
            for (int b = 0; b < BB; ++b)
                if (bb != b) nl += pp[bb * 4 + b];
        out[12288] = cd + 0.1f * (nl / 12.0f);
    }
}

// ---------------- host launcher ----------------
extern "C" void kernel_launch(void* const* d_in, const int* in_sizes, int n_in,
                              void* d_out, int out_size, void* d_ws, size_t ws_size,
                              hipStream_t stream)
{
    (void)in_sizes; (void)n_in; (void)out_size; (void)ws_size;
    const float* x        = (const float*)d_in[0];
    const float* gt       = (const float*)d_in[1];
    const float* features = (const float*)d_in[2];
    auto PW = [&](int l, int k) { return (const float*)d_in[3 + l * 14 + k]; };

    size_t off = 0;
    auto alloc = [&](size_t bytes) -> char* {
        char* p = (char*)d_ws + off;
        off += (bytes + 255) & ~(size_t)255;
        return p;
    };
    float*          f32a   = (float*)alloc((size_t)TOK * EDIM * 4);
    float*          f32b   = (float*)alloc((size_t)TOK * EDIM * 4);
    float*          f32c   = (float*)alloc((size_t)TOK * EDIM * 4);
    unsigned short* bfa    = (unsigned short*)alloc((size_t)TOK * EDIM * 2);
    unsigned short* obf    = (unsigned short*)alloc((size_t)TOK * EDIM * 2);
    unsigned short* bfqkv  = (unsigned short*)alloc((size_t)TOK * QKVN * 2);
    unsigned short* featbf = (unsigned short*)alloc((size_t)TOK * 256 * 2);
    unsigned short* wproj  = (unsigned short*)alloc((size_t)1024 * 1024 * 2);
    unsigned short* win    = (unsigned short*)alloc((size_t)3072 * 1024 * 2);
    unsigned short* wout   = (unsigned short*)alloc((size_t)1024 * 1024 * 2);
    unsigned short* wl11   = (unsigned short*)alloc((size_t)1024 * 1024 * 2);
    unsigned short* wl12   = (unsigned short*)alloc((size_t)1024 * 1024 * 2);
    float*          predws = (float*)alloc((size_t)12288 * 4);
    int*            idxws  = (int*)alloc((size_t)4096 * 10 * 4);
    float*          accs   = (float*)alloc((size_t)64 * 4);

    auto cvt = [&](const float* src, unsigned short* dst, int n) {
        cvt_f32_bf16<<<(n + 255) / 256, 256, 0, stream>>>(src, dst, n);
    };

    // features -> token-major bf16
    feat_to_tok<<<(TOK * 256) / 256, 256, 0, stream>>>(features, featbf);

    for (int l = 0; l < 3; ++l) {
        int cin = (l == 0) ? 256 : 1024;
        const unsigned short* inbf = (l == 0) ? featbf : bfa;

        cvt(PW(l, 0),  wproj, 1024 * cin);
        cvt(PW(l, 4),  win,   3072 * 1024);
        cvt(PW(l, 6),  wout,  1024 * 1024);
        cvt(PW(l, 10), wl11,  1024 * 1024);
        cvt(PW(l, 12), wl12,  1024 * 1024);

        // proj -> f32a   (tiles: (TOK/16) * (N/64) wave-groups, 8 waves/block)
        gemm_wmma<<<(512 * 16) / 8, 256, 0, stream>>>(inbf, wproj, PW(l, 1),
                                                      f32a, nullptr, TOK, 1024, cin, 0);
        // ln13 -> f32b (f32), bfa (bf16)
        ln_kernel<<<TOK, 256, 0, stream>>>(f32a, nullptr, PW(l, 2), PW(l, 3), f32b, bfa);
        // qkv -> bfqkv (bf16)
        gemm_wmma<<<(512 * 48) / 8, 256, 0, stream>>>(bfa, win, PW(l, 5),
                                                      nullptr, bfqkv, TOK, QKVN, 1024, 0);
        // attention -> obf (bf16)
        attn_wmma<<<dim3(128, NH, BB), 32, 0, stream>>>(bfqkv, obf);
        // out proj -> f32c
        gemm_wmma<<<(512 * 16) / 8, 256, 0, stream>>>(obf, wout, PW(l, 7),
                                                      f32c, nullptr, TOK, 1024, 1024, 0);
        // ln12(residual f32b + f32c) -> f32a (f32), bfa (bf16)
        ln_kernel<<<TOK, 256, 0, stream>>>(f32c, f32b, PW(l, 8), PW(l, 9), f32a, bfa);
        // ff1 + GELU -> obf (bf16)
        gemm_wmma<<<(512 * 16) / 8, 256, 0, stream>>>(bfa, wl11, PW(l, 11),
                                                      nullptr, obf, TOK, 1024, 1024, 1);
        // ff2 -> f32c
        gemm_wmma<<<(512 * 16) / 8, 256, 0, stream>>>(obf, wl12, PW(l, 13),
                                                      f32c, nullptr, TOK, 1024, 1024, 0);
        // layer out = f32a + f32c -> f32b (f32), bfa (bf16 for next layer)
        add_cvt<<<(TOK * EDIM) / 256, 256, 0, stream>>>(f32a, f32c, f32b, bfa, TOK * EDIM);
    }

    // softmax over N + pred
    softmax_pred<<<BB * 1024, 256, 0, stream>>>(f32b, x, predws, (float*)d_out);

    // losses
    zero_f<<<1, 32, 0, stream>>>(accs, 32);
    chamfer_min<<<(BB * 1024 + 255) / 256, 256, 0, stream>>>(predws, gt, 1024, 2048,
                                                             accs, 1.0f / 1024.0f);
    chamfer_min<<<(BB * 2048 + 255) / 256, 256, 0, stream>>>(gt, predws, 2048, 1024,
                                                             accs, 1.0f / 2048.0f);
    knn_kernel<<<(BB * 1024 + 255) / 256, 256, 0, stream>>>(predws, idxws);
    neighbor_kernel<<<(BB * 1024 + 255) / 256, 256, 0, stream>>>(predws, idxws, accs + 4);
    finalize_kernel<<<1, 32, 0, stream>>>(accs, accs + 4, (float*)d_out);
}
